// PEMA_82970178224285
// MI455X (gfx1250) — compile-verified, matrix operand order checked by reading, not run on hardware
//
#include <hip/hip_runtime.h>
#include <hip/hip_bf16.h>

#define Bk 4
#define Lk 2048
#define Kk 30
#define HIDk 128
#define NHk 4
#define EPSk 1e-6f
#define NEG_INFk (-3.402823466e38f)

typedef __attribute__((ext_vector_type(16))) __bf16 v16bf;
typedef __attribute__((ext_vector_type(8)))  float  v8f;
typedef __attribute__((ext_vector_type(4)))  unsigned u32x4;
typedef __attribute__((ext_vector_type(8)))  int      i32x8;
typedef __attribute__((ext_vector_type(4)))  int      i32x4;

union AFrag { unsigned u[8]; v16bf v; };

__device__ __forceinline__ unsigned short f2bf_bits(float f) {
    unsigned x = __float_as_uint(f);
    unsigned r = x + 0x7FFFu + ((x >> 16) & 1u);
    return (unsigned short)(r >> 16);
}
__device__ __forceinline__ float bf2f(unsigned short b) {
    return __uint_as_float(((unsigned)b) << 16);
}

__device__ __forceinline__ v8f wmma_bf16(const AFrag& a, const AFrag& b, v8f c) {
    return __builtin_amdgcn_wmma_f32_16x16x32_bf16(false, a.v, false, b.v, (short)0, c, false, false);
}

// A/B fragment (16x32 bf16, row-major source with leading dim ld, ushort elements).
// ISA layout: lane m=lane&15; lanes>=16 get K+8; VGPR 0..3 -> K pairs 0..7(+hi), 4..7 -> 16..23(+hi).
__device__ __forceinline__ AFrag load_frag_u16(const unsigned short* base, int ld, int kbase, int lane) {
    int m  = lane & 15;
    int hi = (lane & 16) ? 8 : 0;
    const unsigned short* row = base + (long)m * ld + kbase + hi;
    AFrag f;
#pragma unroll
    for (int p = 0; p < 4; ++p) f.u[p]     = *(const unsigned*)(row + 2 * p);
#pragma unroll
    for (int p = 0; p < 4; ++p) f.u[4 + p] = *(const unsigned*)(row + 16 + 2 * p);
    return f;
}

__device__ __forceinline__ void store_tile(float* dst, int ldd, v8f acc, int m0, int n0, int lane) {
    int n  = n0 + (lane & 15);
    int mo = (lane & 16) ? 8 : 0;
#pragma unroll
    for (int r = 0; r < 8; ++r) dst[(long)(m0 + r + mo) * ldd + n] = acc[r];
}

__device__ __forceinline__ float wave_sum(float v) {
#pragma unroll
    for (int o = 16; o; o >>= 1) v += __shfl_xor(v, o, 32);
    return v;
}

// ---- Tensor Data Mover: 1D contiguous copy of `nelems` fp32 global -> LDS ----
#if __has_builtin(__builtin_amdgcn_tensor_load_to_lds) && __has_builtin(__builtin_amdgcn_s_wait_tensorcnt)
#define USE_TDM 1
__device__ __forceinline__ void tdm_load_1d_f32(unsigned lds_off, unsigned long long ga, unsigned nelems) {
    // D# group0: count=1 | lds_addr | global_addr[56:0] | type=2
    u32x4 g0;
    g0[0] = 1u;
    g0[1] = lds_off;
    g0[2] = (unsigned)(ga & 0xFFFFFFFFu);
    g0[3] = (unsigned)((ga >> 32) & 0x01FFFFFFu) | (2u << 30);
    // D# group1: data_size=4B; tensor_dim0=tile_dim0=nelems; tensor_dim1=tile_dim1=1; dim0_stride=nelems
    i32x8 g1;
    g1[0] = (int)(2u << 16);                                      // data_size=2 (4 bytes)
    g1[1] = (int)((nelems & 0xFFFFu) << 16);                      // tensor_dim0[15:0]
    g1[2] = (int)(((nelems >> 16) & 0xFFFFu) | (1u << 16));       // tensor_dim0[31:16] | tensor_dim1=1
    g1[3] = (int)((nelems & 0xFFFFu) << 16);                      // tile_dim0 (nelems < 65536)
    g1[4] = 1;                                                    // tile_dim1=1
    g1[5] = (int)nelems;                                          // tensor_dim0_stride low
    g1[6] = 0;
    g1[7] = 0;
    i32x4 z4 = {};
    i32x8 z8 = {};
    // 6-arg form (clang-23 / therock headers): groups 2,3 unused for <=2D tensors.
    __builtin_amdgcn_tensor_load_to_lds(g0, g1, z4, z4, z8, 0);
}
#define TDM_WAIT() __builtin_amdgcn_s_wait_tensorcnt(0)
#else
#define USE_TDM 0
#endif

// ---- fp32 (IN,OUT) -> bf16 transposed (OUT,IN) weight conversion ----
__global__ void convT_kernel(const float* __restrict__ W, unsigned short* __restrict__ out, int IN, int OUT) {
    int t = blockIdx.x * 256 + threadIdx.x;
    if (t < IN * OUT) {
        int o = t / IN, i = t % IN;
        out[t] = f2bf_bits(W[(long)i * OUT + o]);
    }
}

// ---- C[r,0..127] = A[r,0..127] @ W + bias ; 32 rows per block (rows % 32 == 0). ----
__global__ void gemm128_bias_kernel(const float* __restrict__ A, const unsigned short* __restrict__ WT,
                                    const float* __restrict__ bias, float* __restrict__ C) {
    __shared__ float          sF[32 * HIDk];
    __shared__ unsigned short sT[32 * HIDk];
    const int tid = threadIdx.x;
    const int wid = tid >> 5, lane = tid & 31;
    const long m0 = (long)blockIdx.x * 32;
    const float* Ab = A + m0 * HIDk;

#if USE_TDM
    if (wid == 0) tdm_load_1d_f32((unsigned)(unsigned long long)(void*)sF,
                                  (unsigned long long)(const void*)Ab, 32 * HIDk);
    if (wid == 0) TDM_WAIT();
    __syncthreads();
    for (int idx = tid; idx < 32 * HIDk; idx += 256) sT[idx] = f2bf_bits(sF[idx]);
#else
    for (int idx = tid; idx < 32 * HIDk; idx += 256) sT[idx] = f2bf_bits(Ab[idx]);
#endif
    __syncthreads();

    const int n0 = wid * 16;
    v8f acc0 = {}, acc1 = {};
#pragma unroll
    for (int k = 0; k < HIDk; k += 32) {
        AFrag b  = load_frag_u16(WT + (long)n0 * HIDk, HIDk, k, lane);
        AFrag a0 = load_frag_u16(sT, HIDk, k, lane);
        AFrag a1 = load_frag_u16(sT + 16 * HIDk, HIDk, k, lane);
        acc0 = wmma_bf16(a0, b, acc0);
        acc1 = wmma_bf16(a1, b, acc1);
    }
    int n = n0 + (lane & 15);
    float bv = bias[n];
    int mo = (lane & 16) ? 8 : 0;
#pragma unroll
    for (int r = 0; r < 8; ++r) {
        C[(m0 + r + mo) * HIDk + n]      = acc0[r] + bv;
        C[(m0 + 16 + r + mo) * HIDk + n] = acc1[r] + bv;
    }
}

// ---- Attention + residual + LayerNorm per (b,l). One block per node. ----
#define ATTN_SMEM_BYTES (16384 + 9056 * 4)
__global__ void attn_kernel(const float* __restrict__ hVin, const float* __restrict__ hE,
                            const int* __restrict__ Eidx, const float* __restrict__ mask,
                            const unsigned short* __restrict__ WqT, const unsigned short* __restrict__ Wk1T,
                            const unsigned short* __restrict__ Wk2T, const unsigned short* __restrict__ WvT,
                            const unsigned short* __restrict__ WoT,
                            const float* __restrict__ lng, const float* __restrict__ lnb,
                            float* __restrict__ hVout) {
    extern __shared__ char smem[];
    unsigned short* sA = reinterpret_cast<unsigned short*>(smem);       // 32 x 256 bf16 = 16384 B
    float* F    = reinterpret_cast<float*>(smem + 16384);
    float* sHV  = F;              // 128
    float* sQ   = F + 128;        // 128
    float* sP   = F + 256;        // 32*128 : K1 .* K2   (first reused as fp32 h_E staging)
    float* sVh  = F + 4352;       // 32*128
    float* sLog = F + 8448;       // 4*32
    float* sAtt = F + 8576;       // 4*32
    float* sM   = F + 8704;       // 32
    float* sUpd = F + 8736;       // 128
    float* sX   = F + 8864;       // 128
    float* sRed = F + 8992;       // 64
    float* sF   = sP;             // fp32 TDM staging (15360 B <= 16384 B), dead before sP is written

    const int tid  = threadIdx.x;
    const int wid  = tid >> 5, lane = tid & 31;
    const long node = blockIdx.x;                 // b*L + l
    const int  bidx = (int)(node / Lk);
    const int* ei   = Eidx + node * Kk;
    const float mk_self = mask[node];
    const float* eb = hE + node * (long)Kk * HIDk;

    // ---- stage (TDM for contiguous h_E tile, overlapped with gathers) ----
#if USE_TDM
    if (wid == 0) tdm_load_1d_f32((unsigned)(unsigned long long)(void*)sF,
                                  (unsigned long long)(const void*)eb, Kk * HIDk);
#endif
    if (tid < HIDk) sHV[tid] = hVin[node * HIDk + tid];
    for (int idx = tid; idx < Kk * HIDk; idx += 256) {
        int k = idx >> 7, j = idx & 127;
        int nb = ei[k];
        sA[k * 256 + 128 + j] = f2bf_bits(hVin[((long)bidx * Lk + nb) * HIDk + j]);
    }
    for (int idx = tid; idx < 2 * 256; idx += 256) sA[30 * 256 + idx] = 0;   // rows 30,31
    if (tid < Kk) sM[tid] = mk_self * mask[(long)bidx * Lk + ei[tid]];
#if USE_TDM
    if (wid == 0) TDM_WAIT();
    __syncthreads();
    for (int idx = tid; idx < Kk * HIDk; idx += 256) {
        int k = idx >> 7, j = idx & 127;
        sA[k * 256 + j] = f2bf_bits(sF[idx]);
    }
#else
    for (int idx = tid; idx < Kk * HIDk; idx += 256) {
        int k = idx >> 7, j = idx & 127;
        sA[k * 256 + j] = f2bf_bits(eb[idx]);
    }
#endif
    __syncthreads();

    // ---- WMMA: K1, K2 (store elementwise product), Vh ----
    const int n0 = wid * 16;
#pragma unroll
    for (int mt = 0; mt < 2; ++mt) {
        const unsigned short* arow = sA + mt * 16 * 256;
        v8f c1 = {}, c2 = {};
        for (int k = 0; k < HIDk; k += 32) {
            AFrag a1 = load_frag_u16(arow, 256, k, lane);
            AFrag b1 = load_frag_u16(Wk1T + (long)n0 * HIDk, HIDk, k, lane);
            c1 = wmma_bf16(a1, b1, c1);
            AFrag a2 = load_frag_u16(arow, 256, 128 + k, lane);
            AFrag b2 = load_frag_u16(Wk2T + (long)n0 * HIDk, HIDk, k, lane);
            c2 = wmma_bf16(a2, b2, c2);
        }
        {
            int n = n0 + (lane & 15), mo = (lane & 16) ? 8 : 0;
#pragma unroll
            for (int r = 0; r < 8; ++r) sP[(mt * 16 + r + mo) * HIDk + n] = c1[r] * c2[r];
        }
        v8f cv = {};
        for (int k = 0; k < 256; k += 32) {
            AFrag a = load_frag_u16(arow, 256, k, lane);
            AFrag b = load_frag_u16(WvT + (long)n0 * 256, 256, k, lane);
            cv = wmma_bf16(a, b, cv);
        }
        store_tile(sVh, HIDk, cv, mt * 16, n0, lane);
    }

    // ---- Q = h_V @ Wq (1 row; VALU) ----
    if (tid < HIDk) {
        float q = 0.f;
        const unsigned short* wr = WqT + (long)tid * HIDk;
        for (int i = 0; i < HIDk; ++i) q += sHV[i] * bf2f(wr[i]);
        sQ[tid] = q;
    }
    __syncthreads();

    // ---- logits ----
    if (tid < 128) {
        int k = tid & 31, h = tid >> 5;
        if (k < Kk) {
            float s = 0.f;
            for (int d = 0; d < 32; ++d) {
                int j = h * 32 + d;
                s += sQ[j] * sP[k * HIDk + j];
            }
            s *= (1.0f / 32.0f);
            sLog[h * 32 + k] = (sM[k] > 0.f) ? s : NEG_INFk;
        }
    }
    __syncthreads();

    // ---- softmax over k, then attend *= mask ----
    if (tid < NHk) {
        int h = tid;
        float mx = NEG_INFk;
        for (int k = 0; k < Kk; ++k) mx = fmaxf(mx, sLog[h * 32 + k]);
        float sum = 0.f;
        for (int k = 0; k < Kk; ++k) {
            float e = __expf(sLog[h * 32 + k] - mx);
            sAtt[h * 32 + k] = e;
            sum += e;
        }
        float inv = 1.0f / sum;
        for (int k = 0; k < Kk; ++k) sAtt[h * 32 + k] = sM[k] * sAtt[h * 32 + k] * inv;
    }
    __syncthreads();

    // ---- upd = attend @ Vh ----
    if (tid < HIDk) {
        int h = tid >> 5;
        float s = 0.f;
        for (int k = 0; k < Kk; ++k) s += sAtt[h * 32 + k] * sVh[k * HIDk + tid];
        sUpd[tid] = s;
    }
    __syncthreads();

    // ---- dh = upd @ Wo ; residual ----
    if (tid < HIDk) {
        float s = 0.f;
        const unsigned short* wr = WoT + (long)tid * HIDk;
        for (int i = 0; i < HIDk; ++i) s += sUpd[i] * bf2f(wr[i]);
        sX[tid] = sHV[tid] + s;
    }
    __syncthreads();

    // ---- LayerNorm (ddof=1) ----
    if (wid == 0) {
        float x0 = sX[lane], x1 = sX[lane + 32], x2 = sX[lane + 64], x3 = sX[lane + 96];
        float mu = wave_sum(x0 + x1 + x2 + x3) * (1.0f / 128.0f);
        float d0 = x0 - mu, d1 = x1 - mu, d2 = x2 - mu, d3 = x3 - mu;
        float var = wave_sum(d0 * d0 + d1 * d1 + d2 * d2 + d3 * d3) * (1.0f / 127.0f);
        if (lane == 0) { sRed[0] = mu; sRed[1] = sqrtf(var + EPSk); }
    }
    __syncthreads();
    if (tid < HIDk) {
        float mu = sRed[0], sg = sRed[1];
        float y = lng[tid] * (sX[tid] - mu) / (sg + EPSk) + lnb[tid];
        hVout[node * HIDk + tid] = mk_self * y;
    }
}

// ---- Edge MLP: h_E = LN(relu([h_E | h_KVnew | h_V] @ fc1 + b)) , in-place per (b,l). ----
#define EDGE_SMEM_BYTES (24576 + 16384)
__global__ void edge_kernel(const float* __restrict__ hV, const float* __restrict__ hEin,
                            const int* __restrict__ Eidx, const unsigned short* __restrict__ fc1T,
                            const float* __restrict__ fc1b,
                            const float* __restrict__ ng, const float* __restrict__ nb,
                            float* __restrict__ hEout) {
    extern __shared__ char smem[];
    unsigned short* sA = reinterpret_cast<unsigned short*>(smem);   // 32 x 384 bf16 = 24576 B
    float* sY = reinterpret_cast<float*>(smem + 24576);             // 32 x 128 f32
    float* sF = sY;                                                  // fp32 TDM staging (aliased; dead before sY)

    const int tid = threadIdx.x;
    const int wid = tid >> 5, lane = tid & 31;
    const long node = blockIdx.x;
    const int  bidx = (int)(node / Lk);
    const int* ei = Eidx + node * Kk;
    const float* eb = hEin + node * (long)Kk * HIDk;

#if USE_TDM
    if (wid == 0) tdm_load_1d_f32((unsigned)(unsigned long long)(void*)sF,
                                  (unsigned long long)(const void*)eb, Kk * HIDk);
#endif
    for (int idx = tid; idx < Kk * HIDk; idx += 256) {
        int k = idx >> 7, j = idx & 127;
        int nbn = ei[k];
        sA[k * 384 + 128 + j] = f2bf_bits(hV[((long)bidx * Lk + nbn) * HIDk + j]);
    }
    for (int idx = tid; idx < Kk * HIDk; idx += 256) {
        int k = idx >> 7, j = idx & 127;
        sA[k * 384 + 256 + j] = f2bf_bits(hV[node * HIDk + j]);
    }
    for (int idx = tid; idx < 2 * 384; idx += 256) sA[30 * 384 + idx] = 0;
#if USE_TDM
    if (wid == 0) TDM_WAIT();
    __syncthreads();
    for (int idx = tid; idx < Kk * HIDk; idx += 256) {
        int k = idx >> 7, j = idx & 127;
        sA[k * 384 + j] = f2bf_bits(sF[idx]);
    }
#else
    for (int idx = tid; idx < Kk * HIDk; idx += 256) {
        int k = idx >> 7, j = idx & 127;
        sA[k * 384 + j] = f2bf_bits(eb[idx]);
    }
#endif
    __syncthreads();

    const int n0 = wid * 16;
#pragma unroll
    for (int mt = 0; mt < 2; ++mt) {
        const unsigned short* arow = sA + mt * 16 * 384;
        v8f acc = {};
        for (int k = 0; k < 384; k += 32) {
            AFrag a = load_frag_u16(arow, 384, k, lane);
            AFrag b = load_frag_u16(fc1T + (long)n0 * 384, 384, k, lane);
            acc = wmma_bf16(a, b, acc);
        }
        store_tile(sY, HIDk, acc, mt * 16, n0, lane);
    }
    __syncthreads();

    // bias + relu + LayerNorm per edge row (one wave per row)
    for (int k = wid; k < Kk; k += 8) {
        float x[4];
#pragma unroll
        for (int c = 0; c < 4; ++c) {
            int j = lane + 32 * c;
            x[c] = fmaxf(sY[k * HIDk + j] + fc1b[j], 0.f);
        }
        float mu = wave_sum(x[0] + x[1] + x[2] + x[3]) * (1.0f / 128.0f);
        float vs = 0.f;
#pragma unroll
        for (int c = 0; c < 4; ++c) { float d = x[c] - mu; vs += d * d; }
        float sg = sqrtf(wave_sum(vs) * (1.0f / 127.0f) + EPSk);
        float* orow = hEout + (node * Kk + k) * (long)HIDk;
#pragma unroll
        for (int c = 0; c < 4; ++c) {
            int j = lane + 32 * c;
            orow[j] = ng[j] * (x[c] - mu) / (sg + EPSk) + nb[j];
        }
    }
}

extern "C" void kernel_launch(void* const* d_in, const int* in_sizes, int n_in,
                              void* d_out, int out_size, void* d_ws, size_t ws_size,
                              hipStream_t stream) {
    (void)in_sizes; (void)n_in; (void)out_size; (void)ws_size;
    const float* V      = (const float*)d_in[0];
    const float* E      = (const float*)d_in[1];
    const int*   E_idx  = (const int*)d_in[2];
    const float* mask   = (const float*)d_in[3];
    const float* Ws_w   = (const float*)d_in[4];
    const float* Ws_b   = (const float*)d_in[5];
    const float* Wt_w   = (const float*)d_in[6];
    const float* Wt_b   = (const float*)d_in[7];
    const float* Wq     = (const float*)d_in[8];
    const float* Wk1    = (const float*)d_in[9];
    const float* Wk2    = (const float*)d_in[10];
    const float* Wv     = (const float*)d_in[11];
    const float* Wo     = (const float*)d_in[12];
    const float* ln_g   = (const float*)d_in[13];
    const float* ln_b   = (const float*)d_in[14];
    const float* fc1_w  = (const float*)d_in[15];
    const float* fc1_b  = (const float*)d_in[16];
    const float* norm_g = (const float*)d_in[17];
    const float* norm_b = (const float*)d_in[18];

    const long BL = (long)Bk * Lk;          // 8192
    const long ROWS_E = BL * Kk;            // 245760

    size_t off = 0;
    auto take = [&](size_t bytes) -> void* {
        void* p = (char*)d_ws + off;
        off += (bytes + 255) & ~(size_t)255;
        return p;
    };
    float* hVa = (float*)take(BL * HIDk * 4);
    float* hVb = (float*)take(BL * HIDk * 4);
    float* hE  = (float*)take(ROWS_E * HIDk * 4);
    unsigned short* WsT  = (unsigned short*)take(128 * 128 * 2);
    unsigned short* WtT  = (unsigned short*)take(128 * 128 * 2);
    unsigned short* WqT  = (unsigned short*)take(3 * 128 * 128 * 2);
    unsigned short* Wk1T = (unsigned short*)take(3 * 128 * 128 * 2);
    unsigned short* Wk2T = (unsigned short*)take(3 * 128 * 128 * 2);
    unsigned short* WoT  = (unsigned short*)take(3 * 128 * 128 * 2);
    unsigned short* WvT  = (unsigned short*)take(3 * 256 * 128 * 2);
    unsigned short* fc1T = (unsigned short*)take(384 * 128 * 2);

    // ---- weight prep (bf16, transposed) ----
    convT_kernel<<<64, 256, 0, stream>>>(Ws_w, WsT, 128, 128);
    convT_kernel<<<64, 256, 0, stream>>>(Wt_w, WtT, 128, 128);
    for (int l = 0; l < 3; ++l) {
        convT_kernel<<<64, 256, 0, stream>>>(Wq  + l * 16384, WqT  + l * 16384, 128, 128);
        convT_kernel<<<64, 256, 0, stream>>>(Wk1 + l * 16384, Wk1T + l * 16384, 128, 128);
        convT_kernel<<<64, 256, 0, stream>>>(Wk2 + l * 16384, Wk2T + l * 16384, 128, 128);
        convT_kernel<<<64, 256, 0, stream>>>(Wo  + l * 16384, WoT  + l * 16384, 128, 128);
        convT_kernel<<<128, 256, 0, stream>>>(Wv + l * 32768, WvT  + l * 32768, 256, 128);
    }
    convT_kernel<<<192, 256, 0, stream>>>(fc1_w, fc1T, 384, 128);

    // ---- input embeddings (rows % 32 == 0) ----
    gemm128_bias_kernel<<<(int)(BL / 32), 256, 0, stream>>>(V, WsT, Ws_b, hVa);
    gemm128_bias_kernel<<<(int)(ROWS_E / 32), 256, 0, stream>>>(E, WtT, Wt_b, hE);

    // ---- 3 layers (last layer's edge update is dead -> skipped) ----
    float* hin = hVa;
    for (int i = 0; i < 3; ++i) {
        float* hout = (i == 2) ? (float*)d_out : ((i == 0) ? hVb : hVa);
        attn_kernel<<<(int)BL, 256, ATTN_SMEM_BYTES, stream>>>(
            hin, hE, E_idx, mask,
            WqT + i * 16384, Wk1T + i * 16384, Wk2T + i * 16384,
            WvT + i * 32768, WoT + i * 16384,
            ln_g + i * 128, ln_b + i * 128, hout);
        if (i < 2)
            edge_kernel<<<(int)BL, 256, EDGE_SMEM_BYTES, stream>>>(
                hout, hE, E_idx, fc1T, fc1_b, norm_g, norm_b, hE);
        hin = hout;
    }
}
